// AttentionHead_33552284516461
// MI455X (gfx1250) — matrix operand order repaired and checked by reading, not compiled, not run on hardware
//
#include <hip/hip_runtime.h>
#include <hip/hip_bf16.h>
#include <math.h>

// Problem dims (match reference)
#define B_ 16
#define S_ 2048
#define D_ 1024
#define H_ 128

typedef __attribute__((ext_vector_type(16))) __bf16 v16bf;
typedef __attribute__((ext_vector_type(8)))  float  v8f;

union FragBF { v16bf v; unsigned u[8]; };

__device__ __forceinline__ v8f zero8() {
    v8f z = {0.f, 0.f, 0.f, 0.f, 0.f, 0.f, 0.f, 0.f};
    return z;
}

__device__ __forceinline__ unsigned short f32_to_bf16(float f) {
    unsigned u = __float_as_uint(f);
    u += 0x7FFFu + ((u >> 16) & 1u);   // round to nearest even
    return (unsigned short)(u >> 16);
}

__device__ __forceinline__ unsigned pack2_bf16(float lo, float hi) {
    unsigned a = __float_as_uint(lo); a += 0x7FFFu + ((a >> 16) & 1u);
    unsigned b = __float_as_uint(hi); b += 0x7FFFu + ((b >> 16) & 1u);
    return (b & 0xFFFF0000u) | (a >> 16);
}

__device__ __forceinline__ v8f wmma_bf16(const FragBF& a, const FragBF& b, v8f c) {
    return __builtin_amdgcn_wmma_f32_16x16x32_bf16(false, a.v, false, b.v,
                                                   (short)0, c, false, false);
}

// A-fragment K offset for 16-bit 16x32: lanes<16 -> K 0-15; lanes>=16 -> K 16-31
__device__ __forceinline__ int a_koff(int j, int hi) {
    return (hi ? 16 : 0) + 2 * j;
}

// ---------------------------------------------------------------------------
// Kernel 1: convert Wq/Wk/Wv (f32, [D,H]) -> bf16 transposed W^T [3][H][D]
// ---------------------------------------------------------------------------
__global__ void convert_w(const float* __restrict__ Wq, const float* __restrict__ Wk,
                          const float* __restrict__ Wv, unsigned short* __restrict__ wT) {
    int idx = blockIdx.x * blockDim.x + threadIdx.x;
    int m = idx / (D_ * H_);
    int r = idx % (D_ * H_);
    int d = r / H_, h = r % H_;
    const float* W = (m == 0) ? Wq : (m == 1) ? Wk : Wv;
    wT[(size_t)m * H_ * D_ + (size_t)h * D_ + d] = f32_to_bf16(W[(size_t)d * H_ + h]);
}

// ---------------------------------------------------------------------------
// Kernel 2: fused QKV projection.  grid = B*S/128, block = 256 (8 waves).
// Each wave: 16 rows x 128 cols for ALL THREE of q/k/v, so the fp32 input is
// read and converted to bf16 exactly once, feeding 24 WMMAs per K-step.
// q,k stored bf16 [B,S,H]; v stored transposed bf16 [B,H,S].
// ---------------------------------------------------------------------------
__global__ void __launch_bounds__(256) qkv_proj(
        const float* __restrict__ x, const unsigned short* __restrict__ wT_all,
        unsigned short* __restrict__ qout, unsigned short* __restrict__ kout,
        unsigned short* __restrict__ vtout) {
    const int wave = threadIdx.x >> 5;
    const int lane = threadIdx.x & 31;
    const int laneLo = lane & 15;
    const int hi = (lane >> 4) & 1;
    const int g0 = (blockIdx.x * 8 + wave) * 16;   // first global row of this wave

    v8f acc[3][8];
#pragma unroll
    for (int m = 0; m < 3; ++m)
#pragma unroll
        for (int t = 0; t < 8; ++t) acc[m][t] = zero8();

    for (int k0 = 0; k0 < D_; k0 += 32) {
        // A fragment: 16x32 slice of input rows, f32 -> packed bf16 (built once)
        FragBF a;
        const int g = g0 + laneLo;
#pragma unroll
        for (int j = 0; j < 8; ++j) {
            const float* p = x + (size_t)g * D_ + k0 + a_koff(j, hi);
            a.u[j] = pack2_bf16(p[0], p[1]);
        }
        const int klo = k0 + (hi ? 16 : 0);
#pragma unroll
        for (int m = 0; m < 3; ++m) {
            const unsigned short* wT = wT_all + (size_t)m * H_ * D_;
#pragma unroll
            for (int t = 0; t < 8; ++t) {
                FragBF b;
                const int h = t * 16 + laneLo;
#pragma unroll
                for (int j = 0; j < 8; ++j)
                    b.u[j] = *(const unsigned*)(wT + (size_t)h * D_ + klo + 2 * j);
                acc[m][t] = wmma_bf16(a, b, acc[m][t]);
            }
        }
    }

#pragma unroll
    for (int t = 0; t < 8; ++t) {
        const int h = t * 16 + laneLo;
#pragma unroll
        for (int j = 0; j < 8; ++j) {
            const int row = g0 + (hi ? 8 : 0) + j;     // global row (b*S+s)
            qout[(size_t)row * H_ + h] = f32_to_bf16(acc[0][t][j]);
            kout[(size_t)row * H_ + h] = f32_to_bf16(acc[1][t][j]);
            const int bb = row >> 11, s = row & (S_ - 1);
            vtout[((size_t)bb * H_ + h) * S_ + s] = f32_to_bf16(acc[2][t][j]);
        }
    }
}

// ---------------------------------------------------------------------------
// Kernel 3: causal flash attention.  grid = (S/128, B), block = 256 (8 waves).
// Each wave: 16 query rows, online softmax, O accumulated f32 (16x128).
// V fragments are hoisted ahead of the softmax to overlap with exp/shuffle
// VALU work; next K/V tiles are prefetched (global_prefetch_b8).
// ---------------------------------------------------------------------------
__global__ void __launch_bounds__(256) flash_attn(
        const unsigned short* __restrict__ qbf, const unsigned short* __restrict__ kbf,
        const unsigned short* __restrict__ vtb, float* __restrict__ out) {
    __shared__ __align__(16) unsigned short pbuf[8 * 16 * 32];   // per-wave P tile (bf16)

    const int wave = threadIdx.x >> 5;
    const int lane = threadIdx.x & 31;
    const int laneLo = lane & 15;
    const int hi = (lane >> 4) & 1;
    const int b = blockIdx.y;
    const int s0 = blockIdx.x * 128 + wave * 16;     // first query row of wave

    const unsigned short* qb = qbf + (size_t)b * S_ * H_;
    const unsigned short* kb = kbf + (size_t)b * S_ * H_;
    const unsigned short* vb = vtb + (size_t)b * H_ * S_;
    unsigned short* pw = pbuf + wave * 512;

    // Q fragments: 4 chunks of 16x32 along H
    FragBF qf[4];
#pragma unroll
    for (int c = 0; c < 4; ++c)
#pragma unroll
        for (int j = 0; j < 8; ++j)
            qf[c].u[j] = *(const unsigned*)(qb + (size_t)(s0 + laneLo) * H_
                                            + c * 32 + a_koff(j, hi));

    v8f o[8];
    float m[8], l[8];
#pragma unroll
    for (int t = 0; t < 8; ++t) o[t] = zero8();
#pragma unroll
    for (int j = 0; j < 8; ++j) { m[j] = -INFINITY; l[j] = 0.f; }

    const float scale = 0.088388347648318447f;   // 128^-0.5
    const int rowMax = s0 + 15;

    for (int kBase = 0; kBase <= rowMax; kBase += 32) {
        // ---- S = Q K^T for 32 keys (two 16x16 score tiles) ----
        v8f sc0 = zero8(), sc1 = zero8();
        const int key0 = min(kBase + laneLo, S_ - 1);
        const int key1 = min(kBase + 16 + laneLo, S_ - 1);
#pragma unroll
        for (int c = 0; c < 4; ++c) {
            const int klo = c * 32 + (hi ? 16 : 0);
            FragBF kf0, kf1;
#pragma unroll
            for (int j = 0; j < 8; ++j) {
                kf0.u[j] = *(const unsigned*)(kb + (size_t)key0 * H_ + klo + 2 * j);
                kf1.u[j] = *(const unsigned*)(kb + (size_t)key1 * H_ + klo + 2 * j);
            }
            sc0 = wmma_bf16(qf[c], kf0, sc0);
            sc1 = wmma_bf16(qf[c], kf1, sc1);
        }

        // ---- hoisted V fragment loads (overlap with softmax VALU below) ----
        FragBF vf[8];
        const int kvbase = kBase + (hi ? 16 : 0);
#pragma unroll
        for (int t = 0; t < 8; ++t) {
            const int h = t * 16 + laneLo;
#pragma unroll
            for (int j = 0; j < 8; ++j) {
                const int key = min(kvbase + 2 * j, S_ - 2);
                vf[t].u[j] = *(const unsigned*)(vb + (size_t)h * S_ + key);
            }
        }

        // ---- prefetch next 32-key K/V tiles into cache ----
        const int nb = kBase + 32;
        if (nb <= rowMax) {
            const int pk = min(nb + lane, S_ - 1);
            __builtin_prefetch(kb + (size_t)pk * H_, 0, 3);
            __builtin_prefetch(kb + (size_t)pk * H_ + 64, 0, 3);
#pragma unroll
            for (int i = 0; i < 4; ++i)
                __builtin_prefetch(vb + (size_t)(lane + 32 * i) * S_ + nb, 0, 3);
        }

        // ---- scale, causal mask, online softmax (row = per-VGPR-slot) ----
#pragma unroll
        for (int j = 0; j < 8; ++j) {
            const int rloc = (hi ? 8 : 0) + j;
            const int row = s0 + rloc;
            const int c0 = kBase + laneLo, c1 = c0 + 16;
            const float x0 = (c0 <= row) ? sc0[j] * scale : -INFINITY;
            const float x1 = (c1 <= row) ? sc1[j] * scale : -INFINITY;
            float mx = fmaxf(x0, x1);
#pragma unroll
            for (int off = 1; off < 16; off <<= 1)
                mx = fmaxf(mx, __shfl_xor(mx, off, 32));
            const float mnew = fmaxf(m[j], mx);
            const float alpha = __expf(m[j] - mnew);
            const float p0 = __expf(x0 - mnew);
            const float p1 = __expf(x1 - mnew);
            float sum = p0 + p1;
#pragma unroll
            for (int off = 1; off < 16; off <<= 1)
                sum += __shfl_xor(sum, off, 32);
            l[j] = l[j] * alpha + sum;
            m[j] = mnew;
#pragma unroll
            for (int t = 0; t < 8; ++t) o[t][j] *= alpha;
            // stage P (bf16) into per-wave LDS tile for C-frag -> A-frag relayout
            pw[rloc * 32 + laneLo]      = f32_to_bf16(p0);
            pw[rloc * 32 + 16 + laneLo] = f32_to_bf16(p1);
        }

        // ---- reload P as an A fragment (16x32 bf16) from LDS ----
        FragBF pf;
#pragma unroll
        for (int j = 0; j < 8; ++j)
            pf.u[j] = *(const unsigned*)(pw + laneLo * 32 + a_koff(j, hi));

        // ---- O += P V ----
#pragma unroll
        for (int t = 0; t < 8; ++t)
            o[t] = wmma_bf16(pf, vf[t], o[t]);
    }

    // ---- normalize and store f32 output [B,S,H] ----
    float* ob = out + (size_t)b * S_ * H_;
#pragma unroll
    for (int t = 0; t < 8; ++t) {
        const int h = t * 16 + laneLo;
#pragma unroll
        for (int j = 0; j < 8; ++j) {
            const int row = s0 + (hi ? 8 : 0) + j;
            ob[(size_t)row * H_ + h] = o[t][j] / l[j];
        }
    }
}

// ---------------------------------------------------------------------------
extern "C" void kernel_launch(void* const* d_in, const int* in_sizes, int n_in,
                              void* d_out, int out_size, void* d_ws, size_t ws_size,
                              hipStream_t stream) {
    const float* x  = (const float*)d_in[0];
    const float* Wq = (const float*)d_in[1];
    const float* Wk = (const float*)d_in[2];
    const float* Wv = (const float*)d_in[3];

    // ws layout (bf16): W^T[3][H][D] | q[B,S,H] | k[B,S,H] | vT[B,H,S]  (~26 MB)
    unsigned short* ws = (unsigned short*)d_ws;
    unsigned short* wT = ws;
    unsigned short* q  = wT + (size_t)3 * H_ * D_;
    unsigned short* k  = q  + (size_t)B_ * S_ * H_;
    unsigned short* vT = k  + (size_t)B_ * S_ * H_;
    float* out = (float*)d_out;

    convert_w<<<(3 * D_ * H_) / 256, 256, 0, stream>>>(Wq, Wk, Wv, wT);
    qkv_proj<<<(B_ * S_) / 128, 256, 0, stream>>>(x, wT, q, k, vT);
    flash_attn<<<dim3(S_ / 128, B_), 256, 0, stream>>>(q, k, vT, out);
}
